// Attention_45724221833663
// MI455X (gfx1250) — compile-verified, matrix operand order checked by reading, not run on hardware
//
#include <hip/hip_runtime.h>

#define B_   8
#define N_   1024
#define P_   77
#define DIM_ 1024
#define H_   16
#define HD_  64
#define KV_  (P_ + N_)        // 1101
#define LN_EPS_ 1e-5f

typedef unsigned short ushort_t;
typedef __attribute__((ext_vector_type(16))) __bf16          v16bf;
typedef __attribute__((ext_vector_type(8)))  float           v8f;
typedef __attribute__((ext_vector_type(8)))  unsigned short  us8;
typedef __attribute__((ext_vector_type(4)))  unsigned int    u32x4;
typedef __attribute__((ext_vector_type(8)))  int             i32x8;
typedef __attribute__((ext_vector_type(4)))  int             i32x4;

#if defined(__HIP_DEVICE_COMPILE__) && __has_builtin(__builtin_amdgcn_tensor_load_to_lds) && __has_builtin(__builtin_amdgcn_s_wait_tensorcnt)
#define HAS_TDM 1
#else
#define HAS_TDM 0
#endif

// ---------- helpers ----------
__device__ __forceinline__ ushort_t f2bf(float f) {
    unsigned u = __float_as_uint(f);
    u += 0x7FFFu + ((u >> 16) & 1u);     // round-to-nearest-even
    return (ushort_t)(u >> 16);
}
__device__ __forceinline__ float bf2f(ushort_t u) {
    return __uint_as_float(((unsigned)u) << 16);
}

// Load a 16x32 bf16 A/B fragment chunk for one lane.
// p points at (row_base + k_window + (laneHi ? 8 : 0)); second 8 elems at p+16.
__device__ __forceinline__ v16bf load_frag(const ushort_t* p) {
    union { us8 u[2]; v16bf v; } t;
    t.u[0] = *reinterpret_cast<const us8*>(p);
    t.u[1] = *reinterpret_cast<const us8*>(p + 16);
    return t.v;
}

__device__ __forceinline__ v8f wmma_bf16(v16bf a, v16bf b, v8f c) {
    return __builtin_amdgcn_wmma_f32_16x16x32_bf16(
        false, a, false, b, (short)0, c, false, false);
}

__device__ __forceinline__ float rmax16(float v) {
    #pragma unroll
    for (int off = 8; off; off >>= 1) v = fmaxf(v, __shfl_xor(v, off, 16));
    return v;
}
__device__ __forceinline__ float rsum16(float v) {
    #pragma unroll
    for (int off = 8; off; off >>= 1) v += __shfl_xor(v, off, 16);
    return v;
}

#if HAS_TDM
// TDM 2D tile load: rows_valid rows x 32 bf16 from row-major src (row stride
// 1024 elems) into LDS at byte offset lds_off; rows past rows_valid read as 0.
// D# layout per CDNA5 ISA 8.3-8.5 (count=1, type=2, data_size=2B).
// This toolchain's builtin takes 6 args: (g0, g1, g2, g3, g4, cpol).
__device__ __forceinline__ void tdm_load_tile(const ushort_t* gsrc, unsigned lds_off,
                                              int rows_valid) {
    unsigned long long ga = (unsigned long long)gsrc;
    u32x4 g0;
    g0[0] = 1u;                                        // count=1, user-mode load
    g0[1] = lds_off;                                   // lds_addr (bytes)
    g0[2] = (unsigned)(ga & 0xFFFFFFFFu);              // global_addr[31:0]
    g0[3] = (unsigned)((ga >> 32) & 0x01FFFFFFu) | 0x80000000u; // addr[56:32] | type=2
    unsigned rv = (unsigned)rows_valid;
    i32x8 g1;
    g1[0] = (int)(1u << 16);                           // data_size=1 (2 bytes)
    g1[1] = (int)(32u << 16);                          // tensor_dim0=32 (lo16)
    g1[2] = (int)((rv & 0xFFFFu) << 16);               // dim0 hi | tensor_dim1 lo16
    g1[3] = (int)((rv >> 16) | (32u << 16));           // dim1 hi | tile_dim0=32
    g1[4] = 128;                                       // tile_dim1=128, tile_dim2=0
    g1[5] = DIM_;                                      // tensor_dim0_stride=1024
    g1[6] = 0;                                         // stride hi | dim1_stride lo
    g1[7] = 0;
    i32x4 z4 = {};
    i32x8 z8 = {};
    __builtin_amdgcn_tensor_load_to_lds(g0, g1, z4, z4, z8, 0);
}
#endif

// ---------- f32 -> bf16 convert ----------
__global__ __launch_bounds__(256) void cvt_kernel(const float* __restrict__ in,
                                                  ushort_t* __restrict__ out, int n) {
    int base = blockIdx.x * 1024 + threadIdx.x;
    #pragma unroll
    for (int k = 0; k < 4; ++k) {
        int idx = base + k * 256;
        if (idx < n) out[idx] = f2bf(in[idx]);
    }
}

// ---------- generic GEMM: C[M,1024] = A[M,1024] @ W[1024,1024]^T (+bias) ----------
// block tile 128x128, 8 waves, wave tile 64x32 (4x2 frags of 16x16)
// TDM path: double-buffered LDS, tensor DMA issued by wave 0, 1 barrier / K-step.
__global__ __launch_bounds__(256) void gemm_bf16(const ushort_t* __restrict__ A,
                                                 const ushort_t* __restrict__ W,
                                                 const float* __restrict__ bias,
                                                 float* __restrict__ outF,
                                                 ushort_t* __restrict__ outB,
                                                 int M) {
    __shared__ ushort_t As[2][128 * 32];
    __shared__ ushort_t Ws[2][128 * 32];

    const int tid   = threadIdx.x;
    const int lane  = tid & 31;
    const int wave  = tid >> 5;
    const int waveM = wave >> 2;          // 0..1 -> 64 rows each
    const int waveN = wave & 3;           // 0..3 -> 32 cols each
    const int col   = lane & 15;
    const int hi    = (lane & 16) ? 8 : 0;
    const int rowbase = blockIdx.x * 128;
    const int colbase = blockIdx.y * 128;

    v8f zero = {};
    v8f acc[4][2];
    #pragma unroll
    for (int m = 0; m < 4; ++m)
        #pragma unroll
        for (int n = 0; n < 2; ++n) acc[m][n] = zero;

#if HAS_TDM
    int rv = M - rowbase; if (rv > 128) rv = 128;     // TDM zero-pads OOB rows
    const unsigned ldsA0 = (unsigned)(unsigned long long)&As[0][0];
    const unsigned ldsA1 = (unsigned)(unsigned long long)&As[1][0];
    const unsigned ldsW0 = (unsigned)(unsigned long long)&Ws[0][0];
    const unsigned ldsW1 = (unsigned)(unsigned long long)&Ws[1][0];
    if (wave == 0) {
        tdm_load_tile(A + (size_t)rowbase * DIM_, ldsA0, rv);
        tdm_load_tile(W + (size_t)colbase * DIM_, ldsW0, 128);
    }
    for (int kk = 0; kk < DIM_; kk += 32) {
        const int buf = (kk >> 5) & 1;
        if (wave == 0) __builtin_amdgcn_s_wait_tensorcnt(0);
        __syncthreads();                               // tile[buf] ready, tile[buf^1] free
        if (wave == 0 && kk + 32 < DIM_) {
            tdm_load_tile(A + (size_t)rowbase * DIM_ + kk + 32, buf ? ldsA0 : ldsA1, rv);
            tdm_load_tile(W + (size_t)colbase * DIM_ + kk + 32, buf ? ldsW0 : ldsW1, 128);
        }
        v16bf bfr[2];
        #pragma unroll
        for (int n = 0; n < 2; ++n)
            bfr[n] = load_frag(&Ws[buf][(waveN * 32 + n * 16 + col) * 32 + hi]);
        #pragma unroll
        for (int m = 0; m < 4; ++m) {
            v16bf afr = load_frag(&As[buf][(waveM * 64 + m * 16 + col) * 32 + hi]);
            #pragma unroll
            for (int n = 0; n < 2; ++n)
                acc[m][n] = wmma_bf16(afr, bfr[n], acc[m][n]);
        }
    }
#else
    const int lr = tid >> 1;              // 0..127
    const int lc = (tid & 1) * 16;        // 0 / 16
    for (int kk = 0; kk < DIM_; kk += 32) {
        int ar = rowbase + lr;
        us8 a0 = {}, a1 = {};
        if (ar < M) {
            const us8* src = reinterpret_cast<const us8*>(A + (size_t)ar * DIM_ + kk + lc);
            a0 = src[0]; a1 = src[1];
        }
        *reinterpret_cast<us8*>(&As[0][lr * 32 + lc])     = a0;
        *reinterpret_cast<us8*>(&As[0][lr * 32 + lc + 8]) = a1;
        const us8* wsrc = reinterpret_cast<const us8*>(W + (size_t)(colbase + lr) * DIM_ + kk + lc);
        *reinterpret_cast<us8*>(&Ws[0][lr * 32 + lc])     = wsrc[0];
        *reinterpret_cast<us8*>(&Ws[0][lr * 32 + lc + 8]) = wsrc[1];
        __syncthreads();
        v16bf bfr[2];
        #pragma unroll
        for (int n = 0; n < 2; ++n)
            bfr[n] = load_frag(&Ws[0][(waveN * 32 + n * 16 + col) * 32 + hi]);
        #pragma unroll
        for (int m = 0; m < 4; ++m) {
            v16bf afr = load_frag(&As[0][(waveM * 64 + m * 16 + col) * 32 + hi]);
            #pragma unroll
            for (int n = 0; n < 2; ++n)
                acc[m][n] = wmma_bf16(afr, bfr[n], acc[m][n]);
        }
        __syncthreads();
    }
#endif

    #pragma unroll
    for (int m = 0; m < 4; ++m)
        #pragma unroll
        for (int n = 0; n < 2; ++n)
            #pragma unroll
            for (int r = 0; r < 8; ++r) {
                int row  = rowbase + waveM * 64 + m * 16 + r + hi;
                int ccol = colbase + waveN * 32 + n * 16 + col;
                if (row < M) {
                    float v = acc[m][n][r];
                    if (bias) v += bias[ccol];
                    if (outB) outB[(size_t)row * DIM_ + ccol] = f2bf(v);
                    else      outF[(size_t)row * DIM_ + ccol] = v;
                }
            }
}

// ---------- attention: flash-style, gated first column ----------
__global__ __launch_bounds__(256) void attn_kernel(const ushort_t* __restrict__ qb,
                                                   const ushort_t* __restrict__ kself,
                                                   const ushort_t* __restrict__ vself,
                                                   const ushort_t* __restrict__ ktext,
                                                   const ushort_t* __restrict__ vtext,
                                                   const float* __restrict__ gate,
                                                   ushort_t* __restrict__ outb) {
    __shared__ ushort_t Ks[32 * 64];       // [kv][d]
    __shared__ ushort_t Vts[64 * 32];      // [d][kv] (transposed)
    __shared__ ushort_t Ps[8 * 16 * 32];   // per-wave P scratch

    const int tid  = threadIdx.x;
    const int lane = tid & 31;
    const int wave = tid >> 5;
    const int col  = lane & 15;
    const int hi   = (lane & 16) ? 8 : 0;

    const int bid = blockIdx.x;
    const int b   = bid >> 7;              // / (H_*8)
    const int rem = bid & 127;
    const int h   = rem >> 3;
    const int qt  = rem & 7;
    const int qbase = qt * 128;

    // Q fragments (16x64 -> two 16x32 A-frags)
    const ushort_t* qptr = qb + ((size_t)(b * N_ + qbase + wave * 16 + col)) * DIM_ + h * HD_;
    v16bf aQ0 = load_frag(qptr + 0  + hi);
    v16bf aQ1 = load_frag(qptr + 32 + hi);

    v8f O0 = {}, O1 = {}, O2 = {}, O3 = {};
    float m_s[8], l_s[8];
    #pragma unroll
    for (int r = 0; r < 8; ++r) { m_s[r] = -3.0e38f; l_s[r] = 0.0f; }

    const int kvrow = tid >> 3;            // 0..31
    const int chunk = (tid & 7) * 8;       // 0..56

    for (int t = 0; t < 35; ++t) {
        const int kvbase = t * 32;
        // cooperative K / V^T tile load (zero-padded past KV_)
        {
            int kv = kvbase + kvrow;
            us8 kval = {}, vval = {};
            if (kv < P_) {
                size_t off = ((size_t)(b * P_ + kv)) * DIM_ + h * HD_ + chunk;
                kval = *reinterpret_cast<const us8*>(ktext + off);
                vval = *reinterpret_cast<const us8*>(vtext + off);
            } else if (kv < KV_) {
                size_t off = ((size_t)(b * N_ + (kv - P_))) * DIM_ + h * HD_ + chunk;
                kval = *reinterpret_cast<const us8*>(kself + off);
                vval = *reinterpret_cast<const us8*>(vself + off);
            }
            *reinterpret_cast<us8*>(&Ks[kvrow * 64 + chunk]) = kval;
            #pragma unroll
            for (int j = 0; j < 8; ++j) Vts[(chunk + j) * 32 + kvrow] = vval[j];
        }
        __syncthreads();

        // scores: 16x32 (two 16x16 C frags)
        v8f s0 = {}, s1 = {};
        {
            v16bf b00 = load_frag(&Ks[(col) * 64 + 0  + hi]);
            v16bf b01 = load_frag(&Ks[(col) * 64 + 32 + hi]);
            s0 = wmma_bf16(aQ0, b00, s0);
            s0 = wmma_bf16(aQ1, b01, s0);
            v16bf b10 = load_frag(&Ks[(16 + col) * 64 + 0  + hi]);
            v16bf b11 = load_frag(&Ks[(16 + col) * 64 + 32 + hi]);
            s1 = wmma_bf16(aQ0, b10, s1);
            s1 = wmma_bf16(aQ1, b11, s1);
        }

        const int kv0 = kvbase + col;
        const int kv1 = kv0 + 16;
        const bool bad0 = (kv0 == 0) || (kv0 >= KV_);  // first key col gated separately
        const bool bad1 = (kv1 >= KV_);
        #pragma unroll
        for (int r = 0; r < 8; ++r) {
            s0[r] = bad0 ? -3.0e38f : s0[r] * 0.125f;  // 1/sqrt(64)
            s1[r] = bad1 ? -3.0e38f : s1[r] * 0.125f;
        }

        const int wbase = wave * 16 * 32;
        #pragma unroll
        for (int r = 0; r < 8; ++r) {
            float mx   = rmax16(fmaxf(s0[r], s1[r]));
            float mnew = fmaxf(m_s[r], mx);
            float corr = __expf(m_s[r] - mnew);
            m_s[r] = mnew;
            float p0 = __expf(s0[r] - mnew);
            float p1 = __expf(s1[r] - mnew);
            float rs = rsum16(p0 + p1);
            l_s[r] = l_s[r] * corr + rs;
            O0[r] *= corr; O1[r] *= corr; O2[r] *= corr; O3[r] *= corr;
            int prow = r + hi;
            Ps[wbase + prow * 32 + col]      = f2bf(p0);
            Ps[wbase + prow * 32 + 16 + col] = f2bf(p1);
        }
        asm volatile("s_wait_dscnt 0" ::: "memory");   // wave-local LDS RAW

        v16bf aP  = load_frag(&Ps[wbase + col * 32 + hi]);
        v16bf bv0 = load_frag(&Vts[(0  + col) * 32 + hi]);
        v16bf bv1 = load_frag(&Vts[(16 + col) * 32 + hi]);
        v16bf bv2 = load_frag(&Vts[(32 + col) * 32 + hi]);
        v16bf bv3 = load_frag(&Vts[(48 + col) * 32 + hi]);
        O0 = wmma_bf16(aP, bv0, O0);
        O1 = wmma_bf16(aP, bv1, O1);
        O2 = wmma_bf16(aP, bv2, O2);
        O3 = wmma_bf16(aP, bv3, O3);
        __syncthreads();
    }

    // finalize: normalize + gated first-token value, store bf16
    const float g0 = tanhf(gate[h]);
    const size_t v0base = ((size_t)(b * P_)) * DIM_ + h * HD_;
    #pragma unroll
    for (int r = 0; r < 8; ++r) {
        float inv = 1.0f / l_s[r];
        int m = r + hi;
        size_t orow = ((size_t)(b * N_ + qbase + wave * 16 + m)) * DIM_ + h * HD_;
        outb[orow + 0  + col] = f2bf(O0[r] * inv + g0 * bf2f(vtext[v0base + 0  + col]));
        outb[orow + 16 + col] = f2bf(O1[r] * inv + g0 * bf2f(vtext[v0base + 16 + col]));
        outb[orow + 32 + col] = f2bf(O2[r] * inv + g0 * bf2f(vtext[v0base + 32 + col]));
        outb[orow + 48 + col] = f2bf(O3[r] * inv + g0 * bf2f(vtext[v0base + 48 + col]));
    }
}

// ---------- LayerNorm (wave per row) ----------
__global__ __launch_bounds__(256) void ln_kernel(const ushort_t* __restrict__ in,
                                                 const float* __restrict__ g,
                                                 const float* __restrict__ bet,
                                                 ushort_t* __restrict__ out) {
    const int wave = threadIdx.x >> 5;
    const int lane = threadIdx.x & 31;
    const size_t row = (size_t)blockIdx.x * 8 + wave;
    const ushort_t* p = in + row * DIM_ + lane * 32;

    float v[32];
    #pragma unroll
    for (int c = 0; c < 4; ++c) {
        us8 u = *reinterpret_cast<const us8*>(p + c * 8);
        #pragma unroll
        for (int j = 0; j < 8; ++j) v[c * 8 + j] = bf2f(u[j]);
    }
    float s = 0.f, s2 = 0.f;
    #pragma unroll
    for (int i = 0; i < 32; ++i) { s += v[i]; s2 += v[i] * v[i]; }
    #pragma unroll
    for (int off = 16; off; off >>= 1) {
        s  += __shfl_xor(s,  off, 32);
        s2 += __shfl_xor(s2, off, 32);
    }
    const float mean = s * (1.0f / DIM_);
    const float var  = s2 * (1.0f / DIM_) - mean * mean;
    const float rstd = rsqrtf(var + LN_EPS_);

    ushort_t* q = out + row * DIM_ + lane * 32;
    #pragma unroll
    for (int i = 0; i < 32; ++i) {
        int c = lane * 32 + i;
        q[i] = f2bf((v[i] - mean) * rstd * g[c] + bet[c]);
    }
}

// ---------- launch ----------
extern "C" void kernel_launch(void* const* d_in, const int* in_sizes, int n_in,
                              void* d_out, int out_size, void* d_ws, size_t ws_size,
                              hipStream_t stream) {
    (void)in_sizes; (void)n_in; (void)out_size; (void)ws_size;
    const float* x    = (const float*)d_in[0];
    const float* xt   = (const float*)d_in[1];
    const float* Wq   = (const float*)d_in[2];
    const float* Wk   = (const float*)d_in[3];
    const float* Wv   = (const float*)d_in[4];
    const float* gate = (const float*)d_in[5];
    const float* ln_g = (const float*)d_in[6];
    const float* ln_b = (const float*)d_in[7];
    const float* Wp   = (const float*)d_in[8];
    const float* bp   = (const float*)d_in[9];
    float* out = (float*)d_out;

    const size_t nX  = (size_t)B_ * N_ * DIM_;    // 8388608
    const size_t nXT = (size_t)B_ * P_ * DIM_;    // 630784
    const size_t nW  = (size_t)DIM_ * DIM_;       // 1048576

    ushort_t* ws = (ushort_t*)d_ws;
    size_t o = 0;
    ushort_t* xb   = ws + o; o += nX;
    ushort_t* xtb  = ws + o; o += nXT;
    ushort_t* wqb  = ws + o; o += nW;
    ushort_t* wkb  = ws + o; o += nW;
    ushort_t* wvb  = ws + o; o += nW;
    ushort_t* wpb  = ws + o; o += nW;
    ushort_t* qbuf = ws + o; o += nX;
    ushort_t* kbuf = ws + o; o += nX;
    ushort_t* vbuf = ws + o; o += nX;
    ushort_t* ktb  = ws + o; o += nXT;
    ushort_t* vtb  = ws + o; o += nXT;
    ushort_t* attnb = xb;    // x no longer needed after projections
    ushort_t* lnb   = kbuf;  // k no longer needed after attention

    auto cvt = [&](const float* s, ushort_t* d, size_t n) {
        cvt_kernel<<<dim3((unsigned)((n + 1023) / 1024)), 256, 0, stream>>>(s, d, (int)n);
    };
    cvt(x,  xb,  nX);
    cvt(xt, xtb, nXT);
    cvt(Wq, wqb, nW);
    cvt(Wk, wkb, nW);
    cvt(Wv, wvb, nW);
    cvt(Wp, wpb, nW);

    auto gemm = [&](const ushort_t* A, const ushort_t* W, const float* bias,
                    float* oF, ushort_t* oB, int M) {
        gemm_bf16<<<dim3((unsigned)((M + 127) / 128), 8), 256, 0, stream>>>(A, W, bias, oF, oB, M);
    };
    gemm(xb,  wqb, nullptr, nullptr, qbuf, B_ * N_);
    gemm(xb,  wkb, nullptr, nullptr, kbuf, B_ * N_);
    gemm(xb,  wvb, nullptr, nullptr, vbuf, B_ * N_);
    gemm(xtb, wkb, nullptr, nullptr, ktb,  B_ * P_);
    gemm(xtb, wvb, nullptr, nullptr, vtb,  B_ * P_);

    attn_kernel<<<dim3(B_ * H_ * (N_ / 128)), 256, 0, stream>>>(qbuf, kbuf, vbuf, ktb, vtb, gate, attnb);
    ln_kernel<<<dim3((B_ * N_) / 8), 256, 0, stream>>>(attnb, ln_g, ln_b, lnb);
    gemm(lnb, wpb, bp, out, nullptr, B_ * N_);
}
